// CausalAttention_44452911514181
// MI455X (gfx1250) — matrix operand order repaired and checked by reading, not compiled, not run on hardware
//
#include <hip/hip_runtime.h>

// Problem constants (from reference): B=2, N=2048, MEM=2048, J=4096, DIM=1024, H=16, d=64
#define BQ    2
#define NQ    2048
#define MEMQ  2048
#define JQ    4096
#define DIMQ  1024
#define HQ    16
#define DQ    64
#define SCALEQ 0.125f   // 64^-0.5

typedef __attribute__((ext_vector_type(16))) __bf16 v16bf;
typedef __attribute__((ext_vector_type(8)))  float  v8f;

union FragU {
    v16bf v;
    uint4 q[2];
};

// Hardware f32 -> bf16 (RNE); clang selects the native convert on gfx1250.
__device__ __forceinline__ unsigned short f2bf(float f) {
    union { __bf16 h; unsigned short s; } u;
    u.h = (__bf16)f;
    return u.s;
}

__device__ __forceinline__ v8f wmma_bf16(v16bf a, v16bf b, v8f c) {
    return __builtin_amdgcn_wmma_f32_16x16x32_bf16(false, a, false, b, (short)0, c, false, false);
}

// A fragment (16x32 bf16, M x K). Lane L: row = m0 + (L&15); lane-half h = L>>4
// holds K = kb + 8h + [0..7] (elems 0..7) and K = kb + 16 + 8h + [0..7] (elems 8..15).
__device__ __forceinline__ v16bf load_a_frag(const unsigned short* A, int lda,
                                             int m0, int kb, int lane) {
    int row  = m0 + (lane & 15);
    int half = lane >> 4;
    const unsigned short* p = A + (size_t)row * lda + kb + half * 8;
    FragU f;
    f.q[0] = *(const uint4*)(p);
    f.q[1] = *(const uint4*)(p + 16);
    return f.v;
}

// B fragment (32x16 bf16, K x N) sourced from transposed storage BT[N][K].
// Lane L: col = n0 + (L&15); half h holds K = kb + 8h + [0..7] (elems 0..7) and K = kb + 16 + 8h + [0..7] (elems 8..15).
__device__ __forceinline__ v16bf load_b_frag(const unsigned short* BT, int ldb,
                                             int n0, int kb, int lane) {
    int col  = n0 + (lane & 15);
    int half = lane >> 4;
    const unsigned short* p = BT + (size_t)col * ldb + kb + half * 8;
    FragU f;
    f.q[0] = *(const uint4*)(p);
    f.q[1] = *(const uint4*)(p + 16);
    return f.v;
}

// Register-blocked GEMM: one wave computes (16*MT) x (16*NT) output.
// MT=4, NT=2: 8 accum tiles (64 VGPRs) + fragments stays inside the 256-VGPR
// directly-addressable budget with room for the compiler's load pipelining.
template <int MT, int NT>
__device__ __forceinline__ void gemm_block(const unsigned short* A, int lda,
                                           const unsigned short* BT, int ldb,
                                           int m0, int n0, int K, int lane,
                                           v8f acc[MT * NT]) {
    for (int kb = 0; kb < K; kb += 32) {
        v16bf af[MT], bf[NT];
#pragma unroll
        for (int i = 0; i < MT; ++i) af[i] = load_a_frag(A, lda, m0 + 16 * i, kb, lane);
#pragma unroll
        for (int j = 0; j < NT; ++j) bf[j] = load_b_frag(BT, ldb, n0 + 16 * j, kb, lane);
#pragma unroll
        for (int i = 0; i < MT; ++i)
#pragma unroll
            for (int j = 0; j < NT; ++j)
                acc[i * NT + j] = wmma_bf16(af[i], bf[j], acc[i * NT + j]);
    }
}

// ---------------- context = concat(mem, x) -> bf16 (x4 vectorized) ----------------
__global__ void k_cvt_ctx(const float* __restrict__ x, const float* __restrict__ mem,
                          unsigned short* __restrict__ ctx) {
    size_t e = ((size_t)blockIdx.x * blockDim.x + threadIdx.x) * 4;  // over B*J*DIM
    int c = (int)(e % DIMQ);
    size_t t = e / DIMQ;
    int j = (int)(t % JQ);
    int b = (int)(t / JQ);
    const float* src = (j < MEMQ) ? &mem[((size_t)b * MEMQ + j) * DIMQ + c]
                                  : &x[((size_t)b * NQ + (j - MEMQ)) * DIMQ + c];
    float4 v = *(const float4*)src;
    union { uint2 q; unsigned short s[4]; } pk;
    pk.s[0] = f2bf(v.x); pk.s[1] = f2bf(v.y);
    pk.s[2] = f2bf(v.z); pk.s[3] = f2bf(v.w);
    *(uint2*)(ctx + e) = pk.q;
}

// ---------------- W[K][N] f32 -> WT[N][K] bf16 ----------------
__global__ void k_wt(const float* __restrict__ W, unsigned short* __restrict__ WT,
                     int Kd, int Nd) {
    __shared__ float tile[32][33];
    int nt = blockIdx.x * 32, kt = blockIdx.y * 32;
    int tx = threadIdx.x, ty = threadIdx.y;  // 32 x 8
#pragma unroll
    for (int r = 0; r < 32; r += 8)
        tile[ty + r][tx] = W[(size_t)(kt + ty + r) * Nd + nt + tx];
    __syncthreads();
#pragma unroll
    for (int r = 0; r < 32; r += 8)
        WT[(size_t)(nt + ty + r) * Kd + kt + tx] = f2bf(tile[tx][ty + r]);
}

// ---------------- q = x @ Wq  (bf16 out, row-major [b][i][dim]) ----------------
__global__ void __launch_bounds__(256)
k_proj_q(const unsigned short* __restrict__ ctx,
         const unsigned short* __restrict__ WqT,
         unsigned short* __restrict__ qb) {
    int lane = threadIdx.x & 31, wv = threadIdx.x >> 5;
    int n0 = (blockIdx.x * 8 + wv) * 32;
    int m0 = blockIdx.y * 64;
    int b  = blockIdx.z;
    const unsigned short* A = ctx + ((size_t)b * JQ + MEMQ) * DIMQ;  // x rows
    v8f acc[8] = {};
    gemm_block<4, 2>(A, DIMQ, WqT, DIMQ, m0, n0, DIMQ, lane, acc);
    unsigned short* O = qb + (size_t)b * NQ * DIMQ;
    int colbase = lane & 15, half = lane >> 4;
#pragma unroll
    for (int i = 0; i < 4; ++i)
#pragma unroll
        for (int j = 0; j < 2; ++j)
#pragma unroll
            for (int r = 0; r < 8; ++r)
                O[(size_t)(m0 + 16 * i + r + 8 * half) * DIMQ + n0 + 16 * j + colbase] =
                    f2bf(acc[i * 2 + j][r]);
}

// ------- kv = context @ Wkv : k -> [b][j][dim] bf16, v -> vT [b][h][d][j] bf16 -------
__global__ void __launch_bounds__(256)
k_proj_kv(const unsigned short* __restrict__ ctx,
          const unsigned short* __restrict__ WkvT,
          unsigned short* __restrict__ kb,
          unsigned short* __restrict__ vT) {
    int lane = threadIdx.x & 31, wv = threadIdx.x >> 5;
    int n0 = (blockIdx.x * 8 + wv) * 32;   // 32-aligned; K/V split is wave-uniform
    int m0 = blockIdx.y * 64;              // j tile
    int b  = blockIdx.z;
    const unsigned short* A = ctx + (size_t)b * JQ * DIMQ;
    v8f acc[8] = {};
    gemm_block<4, 2>(A, DIMQ, WkvT, DIMQ, m0, n0, DIMQ, lane, acc);
    int colbase = lane & 15, half = lane >> 4;
    if (n0 < DIMQ) {  // K half
        unsigned short* O = kb + (size_t)b * JQ * DIMQ;
#pragma unroll
        for (int i = 0; i < 4; ++i)
#pragma unroll
            for (int j = 0; j < 2; ++j)
#pragma unroll
                for (int r = 0; r < 8; ++r)
                    O[(size_t)(m0 + 16 * i + r + 8 * half) * DIMQ + n0 + 16 * j + colbase] =
                        f2bf(acc[i * 2 + j][r]);
    } else {          // V half: transposed store; 8 rows are j-contiguous -> one b128
        int h = (n0 - DIMQ) >> 6;
        unsigned short* O = vT + (((size_t)b * HQ + h) * DQ) * JQ;  // [dc][j]
#pragma unroll
        for (int i = 0; i < 4; ++i)
#pragma unroll
            for (int j = 0; j < 2; ++j) {
                int dc = (n0 - DIMQ + 16 * j + colbase) & 63;
                union { uint4 q; unsigned short s[8]; } pk;
#pragma unroll
                for (int r = 0; r < 8; ++r) pk.s[r] = f2bf(acc[i * 2 + j][r]);
                *(uint4*)(O + (size_t)dc * JQ + m0 + 16 * i + 8 * half) = pk.q;
            }
    }
}

// ---------------- flash attention, one wave = 16 query rows ----------------
#define PSTR 40  // LDS row stride (ushorts): 80B, 16B aligned
__global__ void __launch_bounds__(128)
k_attn(const unsigned short* __restrict__ qb,
       const unsigned short* __restrict__ kb,
       const unsigned short* __restrict__ vT,
       const float* __restrict__ expire,
       unsigned short* __restrict__ ob) {
    __shared__ unsigned short Plds[4 * 16 * PSTR];
    int lane = threadIdx.x & 31, wv = threadIdx.x >> 5;
    int i0 = (blockIdx.x * 4 + wv) * 16;
    int h  = blockIdx.y;
    int b  = blockIdx.z;

    const unsigned short* Q  = qb + (size_t)b * NQ * DIMQ + h * DQ;
    const unsigned short* Km = kb + (size_t)b * JQ * DIMQ + h * DQ;
    const unsigned short* V  = vT + (((size_t)b * HQ + h) * DQ) * JQ;
    const float* em = expire + (size_t)b * JQ;
    unsigned short* Pl = Plds + wv * 16 * PSTR;

    v16bf qf0 = load_a_frag(Q, DIMQ, i0, 0, lane);
    v16bf qf1 = load_a_frag(Q, DIMQ, i0, 32, lane);

    int colbase = lane & 15, half = lane >> 4;
    float mrow[8], lrow[8];
#pragma unroll
    for (int r = 0; r < 8; ++r) { mrow[r] = -3.0e38f; lrow[r] = 0.0f; }
    v8f o0 = {}, o1 = {}, o2 = {}, o3 = {};

    int jend = i0 + 16 + MEMQ;          // first fully-masked column +1, padded
    if (jend > JQ) jend = JQ;
    jend = (jend + 31) & ~31;

    for (int jt = 0; jt < jend; jt += 32) {
        if (jt + 32 < jend)
            __builtin_prefetch(Km + (size_t)(jt + 32 + colbase) * DIMQ, 0, 0);

        // S = Q K^T  (two 16-col tiles)
        v8f s[2];
#pragma unroll
        for (int t = 0; t < 2; ++t) {
            v8f acc = {};
            acc = wmma_bf16(qf0, load_b_frag(Km, DIMQ, jt + 16 * t, 0, lane), acc);
            acc = wmma_bf16(qf1, load_b_frag(Km, DIMQ, jt + 16 * t, 32, lane), acc);
            s[t] = acc;
        }
        // scale + causal mask + tile row-max (reduce across 16 lanes of this half)
        float tmax[8];
#pragma unroll
        for (int r = 0; r < 8; ++r) {
            int i = i0 + r + 8 * half;
            float s0 = s[0][r] * SCALEQ;
            float s1 = s[1][r] * SCALEQ;
            if (jt + colbase      > i + MEMQ) s0 = -3.0e38f;
            if (jt + 16 + colbase > i + MEMQ) s1 = -3.0e38f;
            s[0][r] = s0; s[1][r] = s1;
            float mx = fmaxf(s0, s1);
#pragma unroll
            for (int d = 1; d < 16; d <<= 1) mx = fmaxf(mx, __shfl_xor(mx, d));
            tmax[r] = mx;
        }
        float alpha[8];
#pragma unroll
        for (int r = 0; r < 8; ++r) {
            float nm = fmaxf(mrow[r], tmax[r]);
            alpha[r] = __expf(mrow[r] - nm);
            mrow[r] = nm;
        }
        // exponentiate; denominator uses UNGATED exp (softmax first, gate after)
#pragma unroll
        for (int r = 0; r < 8; ++r) {
            float e0 = __expf(s[0][r] - mrow[r]);
            float e1 = __expf(s[1][r] - mrow[r]);
            s[0][r] = e0; s[1][r] = e1;
            float sm = e0 + e1;
#pragma unroll
            for (int d = 1; d < 16; d <<= 1) sm += __shfl_xor(sm, d);
            lrow[r] = lrow[r] * alpha[r] + sm;
        }
        // expire gating (numerator only), P -> bf16 -> LDS [16 x 32]
        float em0 = em[jt + colbase];
        float em1 = em[jt + 16 + colbase];
#pragma unroll
        for (int r = 0; r < 8; ++r) {
            int m = r + 8 * half;
            Pl[m * PSTR + colbase]      = f2bf(s[0][r] * em0);
            Pl[m * PSTR + 16 + colbase] = f2bf(s[1][r] * em1);
        }
        asm volatile("s_wait_dscnt 0x0" ::: "memory");  // cross-lane LDS RAW

        // A fragment of P from LDS (16x32, kb=0, lda=PSTR)
        FragU pf;
        {
            const unsigned short* pp = Pl + (size_t)colbase * PSTR + half * 8;
            pf.q[0] = *(const uint4*)(pp);
            pf.q[1] = *(const uint4*)(pp + 16);
        }
        // rescale O then O += P @ V  (V^T rows are d-columns; K=j contiguous)
#pragma unroll
        for (int r = 0; r < 8; ++r) {
            o0[r] *= alpha[r]; o1[r] *= alpha[r];
            o2[r] *= alpha[r]; o3[r] *= alpha[r];
        }
        o0 = wmma_bf16(pf.v, load_b_frag(V, JQ, 0,  jt, lane), o0);
        o1 = wmma_bf16(pf.v, load_b_frag(V, JQ, 16, jt, lane), o1);
        o2 = wmma_bf16(pf.v, load_b_frag(V, JQ, 32, jt, lane), o2);
        o3 = wmma_bf16(pf.v, load_b_frag(V, JQ, 48, jt, lane), o3);
    }

    // normalize and store as bf16 [b*N + i][h*64 + dc]
    unsigned short* O = ob + (size_t)b * NQ * DIMQ + h * DQ;
#pragma unroll
    for (int r = 0; r < 8; ++r) {
        float inv = 1.0f / lrow[r];
        size_t row = (size_t)(i0 + r + 8 * half) * DIMQ;
        O[row + 0  + colbase] = f2bf(o0[r] * inv);
        O[row + 16 + colbase] = f2bf(o1[r] * inv);
        O[row + 32 + colbase] = f2bf(o2[r] * inv);
        O[row + 48 + colbase] = f2bf(o3[r] * inv);
    }
}

// ---------------- out = attn_out @ Wo + bo  (f32) ----------------
__global__ void __launch_bounds__(256)
k_proj_o(const unsigned short* __restrict__ obuf,
         const unsigned short* __restrict__ WoT,
         const float* __restrict__ bo,
         float* __restrict__ out) {
    int lane = threadIdx.x & 31, wv = threadIdx.x >> 5;
    int n0 = (blockIdx.x * 8 + wv) * 32;
    int m0 = blockIdx.y * 64;   // over B*N folded
    v8f acc[8] = {};
    gemm_block<4, 2>(obuf, DIMQ, WoT, DIMQ, m0, n0, DIMQ, lane, acc);
    int colbase = lane & 15, half = lane >> 4;
#pragma unroll
    for (int j = 0; j < 2; ++j) {
        float bias = bo[n0 + 16 * j + colbase];
#pragma unroll
        for (int i = 0; i < 4; ++i)
#pragma unroll
            for (int r = 0; r < 8; ++r)
                out[(size_t)(m0 + 16 * i + r + 8 * half) * DIMQ + n0 + 16 * j + colbase] =
                    acc[i * 2 + j][r] + bias;
    }
}

extern "C" void kernel_launch(void* const* d_in, const int* in_sizes, int n_in,
                              void* d_out, int out_size, void* d_ws, size_t ws_size,
                              hipStream_t stream) {
    const float* x   = (const float*)d_in[0];
    const float* mem = (const float*)d_in[1];
    const float* exm = (const float*)d_in[2];
    const float* Wq  = (const float*)d_in[3];
    const float* Wkv = (const float*)d_in[4];
    const float* Wo  = (const float*)d_in[5];
    const float* bo  = (const float*)d_in[6];
    float* out = (float*)d_out;

    // workspace carve-out (all bf16 buffers as ushort), 256B aligned
    char* ws = (char*)d_ws;
    size_t off = 0;
    auto carve = [&](size_t bytes) { char* p = ws + off; off += (bytes + 255) & ~(size_t)255; return p; };
    unsigned short* ctxb = (unsigned short*)carve((size_t)BQ * JQ * DIMQ * 2);   // 16 MB
    unsigned short* qb   = (unsigned short*)carve((size_t)BQ * NQ * DIMQ * 2);   //  8 MB
    unsigned short* kbuf = (unsigned short*)carve((size_t)BQ * JQ * DIMQ * 2);   // 16 MB
    unsigned short* vT   = (unsigned short*)carve((size_t)BQ * HQ * DQ * JQ * 2);// 16 MB
    unsigned short* obuf = (unsigned short*)carve((size_t)BQ * NQ * DIMQ * 2);   //  8 MB
    unsigned short* WqT  = (unsigned short*)carve((size_t)DIMQ * DIMQ * 2);      //  2 MB
    unsigned short* WkvT = (unsigned short*)carve((size_t)2 * DIMQ * DIMQ * 2);  //  4 MB
    unsigned short* WoT  = (unsigned short*)carve((size_t)DIMQ * DIMQ * 2);      //  2 MB

    // 1) bf16 context (concat) + transposed bf16 weights
    {
        size_t tot4 = (size_t)BQ * JQ * DIMQ / 4;
        k_cvt_ctx<<<dim3((unsigned)(tot4 / 256)), dim3(256), 0, stream>>>(x, mem, ctxb);
        k_wt<<<dim3(DIMQ / 32, DIMQ / 32), dim3(32, 8), 0, stream>>>(Wq, WqT, DIMQ, DIMQ);
        k_wt<<<dim3(2 * DIMQ / 32, DIMQ / 32), dim3(32, 8), 0, stream>>>(Wkv, WkvT, DIMQ, 2 * DIMQ);
        k_wt<<<dim3(DIMQ / 32, DIMQ / 32), dim3(32, 8), 0, stream>>>(Wo, WoT, DIMQ, DIMQ);
    }
    // 2) projections (8 waves/block, one 64x32 register-blocked tile per wave)
    k_proj_q<<<dim3(DIMQ / 32 / 8, NQ / 64, BQ), dim3(256), 0, stream>>>(ctxb, WqT, qb);
    k_proj_kv<<<dim3(2 * DIMQ / 32 / 8, JQ / 64, BQ), dim3(256), 0, stream>>>(ctxb, WkvT, kbuf, vT);
    // 3) flash attention with expire gating (one wave = 16 query rows)
    k_attn<<<dim3(NQ / 16 / 4, HQ, BQ), dim3(128), 0, stream>>>(qb, kbuf, vT, exm, obuf);
    // 4) output projection + bias
    k_proj_o<<<dim3(DIMQ / 32 / 8, BQ * NQ / 64, 1), dim3(256), 0, stream>>>(obuf, WoT, bo, out);
}
